// RWRLayer_25022479466558
// MI455X (gfx1250) — compile-verified, matrix operand order checked by reading, not run on hardware
//
#include <hip/hip_runtime.h>

#define NN   8192
#define FIN  1024
#define FOUT 512

typedef __attribute__((ext_vector_type(16))) __bf16          v16bf;
typedef __attribute__((ext_vector_type(8)))  float           v8f;
typedef __attribute__((ext_vector_type(4)))  float           f32x4;
typedef __attribute__((ext_vector_type(4)))  unsigned int    u32x4;
typedef __attribute__((ext_vector_type(8)))  unsigned short  u16x8;
typedef __attribute__((ext_vector_type(4)))  unsigned short  u16x4;
typedef __attribute__((ext_vector_type(4)))  int             i32x4;

union Frag {
    v16bf          v;
    unsigned short u[16];
    u32x4          q[2];
};

__device__ __forceinline__ unsigned short f2bf(float x) {
    unsigned int u = __float_as_uint(x);
    u += 0x7FFFu + ((u >> 16) & 1u);           // round-to-nearest-even
    return (unsigned short)(u >> 16);
}

__device__ __forceinline__ void ldfrag(Frag& f, const unsigned short* p) {
    f.q[0] = *(const u32x4*)(p);
    f.q[1] = *(const u32x4*)(p + 16);
}

// ---------------------------------------------------------------------------
// K0: f32 -> bf16 conversion. Source is read once -> non-temporal loads;
// bf16 destination stays regular-temporal so it is L2-resident for the GEMM.
// ---------------------------------------------------------------------------
__global__ void cvt_f32_bf16(const float* __restrict__ src,
                             unsigned short* __restrict__ dst, int n) {
    int i = (blockIdx.x * blockDim.x + threadIdx.x) * 4;
    if (i + 3 < n) {
        f32x4 v = __builtin_nontemporal_load((const f32x4*)(src + i));
        u16x4 o;
        o[0] = f2bf(v[0]); o[1] = f2bf(v[1]);
        o[2] = f2bf(v[2]); o[3] = f2bf(v[3]);
        *(u16x4*)(dst + i) = o;
    }
}

// ---------------------------------------------------------------------------
// K1: seq_fts = x @ seq_W^T  (bf16 WMMA, f32 accum).
// 8 waves/block; wave computes 16(M) x 64(N); distance-1 pipelined K loop.
// Writes seq_fts f32 [NN,FOUT] and transposed bf16 sbfT [FOUT,NN].
// ---------------------------------------------------------------------------
__global__ void gemm_seqfts(const unsigned short* __restrict__ xbf,
                            const unsigned short* __restrict__ wbf,
                            float* __restrict__ sf,
                            unsigned short* __restrict__ sbfT) {
    const int lane = threadIdx.x & 31;
    const int wave = threadIdx.x >> 5;
    const int lm   = lane & 15;      // row/col within tile
    const int hk   = lane >> 4;      // K-half selector

    const int gw = blockIdx.x * 8 + wave;          // 4096 wave-tiles
    const int m0 = (gw >> 3) << 4;                 // 512 M-tiles of 16
    const int f0 = (gw & 7) << 6;                  // 8 N-groups of 64

    const unsigned short* arow = xbf + (size_t)(m0 + lm) * FIN + hk * 8;
    const unsigned short* b0p  = wbf + (size_t)(f0 + 0 * 16 + lm) * FIN + hk * 8;
    const unsigned short* b1p  = wbf + (size_t)(f0 + 1 * 16 + lm) * FIN + hk * 8;
    const unsigned short* b2p  = wbf + (size_t)(f0 + 2 * 16 + lm) * FIN + hk * 8;
    const unsigned short* b3p  = wbf + (size_t)(f0 + 3 * 16 + lm) * FIN + hk * 8;

    v8f acc[4];
#pragma unroll
    for (int t = 0; t < 4; ++t)
#pragma unroll
        for (int r = 0; r < 8; ++r) acc[t][r] = 0.0f;

    // prologue
    Frag A, B0;
    ldfrag(A,  arow);
    ldfrag(B0, b0p);

#pragma unroll 2
    for (int kk = 0; kk < FIN - 32; kk += 32) {
        Frag B1, B2, B3, An, Bn;
        ldfrag(B1, b1p + kk);
        acc[0] = __builtin_amdgcn_wmma_f32_16x16x32_bf16(
            false, A.v, false, B0.v, (short)0, acc[0], false, false);
        ldfrag(B2, b2p + kk);
        acc[1] = __builtin_amdgcn_wmma_f32_16x16x32_bf16(
            false, A.v, false, B1.v, (short)0, acc[1], false, false);
        ldfrag(B3, b3p + kk);
        acc[2] = __builtin_amdgcn_wmma_f32_16x16x32_bf16(
            false, A.v, false, B2.v, (short)0, acc[2], false, false);
        ldfrag(An, arow + kk + 32);
        ldfrag(Bn, b0p + kk + 32);
        __builtin_prefetch(arow + kk + 96, 0, 0);
        acc[3] = __builtin_amdgcn_wmma_f32_16x16x32_bf16(
            false, A.v, false, B3.v, (short)0, acc[3], false, false);
        A  = An;
        B0 = Bn;
    }
    {   // epilogue tile: kk = FIN - 32
        const int kk = FIN - 32;
        Frag B1, B2, B3;
        ldfrag(B1, b1p + kk);
        acc[0] = __builtin_amdgcn_wmma_f32_16x16x32_bf16(
            false, A.v, false, B0.v, (short)0, acc[0], false, false);
        ldfrag(B2, b2p + kk);
        acc[1] = __builtin_amdgcn_wmma_f32_16x16x32_bf16(
            false, A.v, false, B1.v, (short)0, acc[1], false, false);
        ldfrag(B3, b3p + kk);
        acc[2] = __builtin_amdgcn_wmma_f32_16x16x32_bf16(
            false, A.v, false, B2.v, (short)0, acc[2], false, false);
        acc[3] = __builtin_amdgcn_wmma_f32_16x16x32_bf16(
            false, A.v, false, B3.v, (short)0, acc[3], false, false);
    }

#pragma unroll
    for (int t = 0; t < 4; ++t) {
        const int nn = f0 + t * 16 + lm;
        u16x8 pk;
#pragma unroll
        for (int r = 0; r < 8; ++r) {
            const int m = m0 + hk * 8 + r;
            sf[(size_t)m * FOUT + nn] = acc[t][r];
            pk[r] = f2bf(acc[t][r]);
        }
        *(u16x8*)(sbfT + (size_t)nn * NN + m0 + hk * 8) = pk;
    }
}

// ---------------------------------------------------------------------------
// K2: f1[i] = seq_fts[i,:].f1_w + f1_b ; f2 likewise. One block per row.
// ---------------------------------------------------------------------------
__global__ void f12_kernel(const float* __restrict__ sf,
                           const float* __restrict__ f1w, const float* __restrict__ f1b,
                           const float* __restrict__ f2w, const float* __restrict__ f2b,
                           float* __restrict__ f1, float* __restrict__ f2) {
    const int i = blockIdx.x;
    float s1 = 0.f, s2 = 0.f;
    for (int c = threadIdx.x; c < FOUT; c += blockDim.x) {
        const float v = sf[(size_t)i * FOUT + c];
        s1 += v * f1w[c];
        s2 += v * f2w[c];
    }
#pragma unroll
    for (int o = 16; o; o >>= 1) {
        s1 += __shfl_down(s1, o, 32);
        s2 += __shfl_down(s2, o, 32);
    }
    __shared__ float r1[8], r2[8];
    if ((threadIdx.x & 31) == 0) {
        r1[threadIdx.x >> 5] = s1;
        r2[threadIdx.x >> 5] = s2;
    }
    __syncthreads();
    if (threadIdx.x == 0) {
        float a = 0.f, b = 0.f;
#pragma unroll
        for (int w = 0; w < 8; ++w) { a += r1[w]; b += r2[w]; }
        f1[i] = a + f1b[0];
        f2[i] = b + f2b[0];
    }
}

// ---------------------------------------------------------------------------
// K3: fused leakyrelu(f1+f2)+adj -> online softmax -> P @ seq_fts (WMMA)
//     -> /l + bias -> ELU -> d_out.
// Block = 16 rows, 8 waves. Producer/consumer over a 256-wide j panel.
// adj is a 256MB one-shot stream -> non-temporal loads (preserve L2 for
// sbfT/wbf reuse); h_prime output is write-once -> non-temporal stores.
// ---------------------------------------------------------------------------
__global__ void flash_attn(const float* __restrict__ adj,
                           const float* __restrict__ f1,
                           const float* __restrict__ f2,
                           const unsigned short* __restrict__ sbfT,
                           const float* __restrict__ bias,
                           float* __restrict__ out) {
    __shared__ unsigned short pbuf[8 * 32 * 16];  // 8 KB: 8 P tiles, lane-major
    __shared__ float          smax[8 * 16];
    __shared__ float          spsum[8 * 16];

    const int lane = threadIdx.x & 31;
    const int wave = threadIdx.x >> 5;
    const int lm   = lane & 15;
    const int hk   = lane >> 4;
    const int i0   = blockIdx.x * 16;
    const int f0   = wave * 64;

    const float* adjrow = adj + (size_t)(i0 + lm) * NN;
    const float  f1v    = f1[i0 + lm];

    const unsigned short* bbase[4];
#pragma unroll
    for (int t = 0; t < 4; ++t)
        bbase[t] = sbfT + (size_t)(f0 + t * 16 + lm) * NN + hk * 8;

    v8f acc[4];
#pragma unroll
    for (int t = 0; t < 4; ++t)
#pragma unroll
        for (int r = 0; r < 8; ++r) acc[t][r] = 0.0f;

    float m_run = -1e30f, l_run = 0.0f;   // per lane, row = lm (dup over hk)

    for (int J = 0; J < NN; J += 256) {
        const int j0 = J + wave * 32;     // this wave's producer subtile

        // ---- phase A: pre-activation tile + row max --------------------
        f32x4 a0 = __builtin_nontemporal_load((const f32x4*)(adjrow + j0 + hk * 8));
        f32x4 a1 = __builtin_nontemporal_load((const f32x4*)(adjrow + j0 + hk * 8 + 4));
        f32x4 a2 = __builtin_nontemporal_load((const f32x4*)(adjrow + j0 + 16 + hk * 8));
        f32x4 a3 = __builtin_nontemporal_load((const f32x4*)(adjrow + j0 + 16 + hk * 8 + 4));
        f32x4 b0 = *(const f32x4*)(f2 + j0 + hk * 8);
        f32x4 b1 = *(const f32x4*)(f2 + j0 + hk * 8 + 4);
        f32x4 b2 = *(const f32x4*)(f2 + j0 + 16 + hk * 8);
        f32x4 b3 = *(const f32x4*)(f2 + j0 + 16 + hk * 8 + 4);
        __builtin_prefetch(adjrow + j0 + 256, 0, 0);

        float pv[16];
#pragma unroll
        for (int p = 0; p < 4; ++p) {
            float lg;
            lg = f1v + b0[p]; lg = lg > 0.f ? lg : 0.2f * lg; pv[p]      = lg + a0[p];
            lg = f1v + b1[p]; lg = lg > 0.f ? lg : 0.2f * lg; pv[p + 4]  = lg + a1[p];
            lg = f1v + b2[p]; lg = lg > 0.f ? lg : 0.2f * lg; pv[p + 8]  = lg + a2[p];
            lg = f1v + b3[p]; lg = lg > 0.f ? lg : 0.2f * lg; pv[p + 12] = lg + a3[p];
        }
        float tmax = pv[0];
#pragma unroll
        for (int p = 1; p < 16; ++p) tmax = fmaxf(tmax, pv[p]);
        tmax = fmaxf(tmax, __shfl_xor(tmax, 16, 32));
        if (hk == 0) smax[wave * 16 + lm] = tmax;
        __syncthreads();                               // (1) smax visible

        // ---- phase B: block-wide max merge -----------------------------
        float m_new = m_run;
#pragma unroll
        for (int w2 = 0; w2 < 8; ++w2) m_new = fmaxf(m_new, smax[w2 * 16 + lm]);
        const float scl = __expf(m_run - m_new);

        // ---- phase C: exponentials -> bf16 P tile into LDS -------------
        Frag P;
        float psum = 0.0f;
#pragma unroll
        for (int p = 0; p < 16; ++p) {
            const float e = __expf(pv[p] - m_new);
            psum += e;
            P.u[p] = f2bf(e);
        }
        psum += __shfl_xor(psum, 16, 32);
        if (hk == 0) spsum[wave * 16 + lm] = psum;
        {
            unsigned short* pw = pbuf + (wave * 32 + lane) * 16;
            *(u32x4*)(pw)     = P.q[0];
            *(u32x4*)(pw + 8) = P.q[1];
        }
        __syncthreads();                               // (2) pbuf/spsum visible

        // ---- phase D: l update + accumulator rescale (once per panel) --
        float tsum = 0.0f;
#pragma unroll
        for (int w2 = 0; w2 < 8; ++w2) tsum += spsum[w2 * 16 + lm];
        l_run = l_run * scl + tsum;
        m_run = m_new;
        float s8[8];
#pragma unroll
        for (int r = 0; r < 8; ++r) s8[r] = __shfl(scl, hk * 8 + r, 32);
#pragma unroll
        for (int t = 0; t < 4; ++t)
#pragma unroll
            for (int r = 0; r < 8; ++r) acc[t][r] *= s8[r];

        // ---- phase E: consume all 8 P tiles, 32 WMMAs ------------------
#pragma unroll
        for (int w2 = 0; w2 < 8; ++w2) {
            Frag P2;
            const unsigned short* pr = pbuf + (w2 * 32 + lane) * 16;
            P2.q[0] = *(const u32x4*)(pr);
            P2.q[1] = *(const u32x4*)(pr + 8);
            const int jj = J + w2 * 32;
#pragma unroll
            for (int t = 0; t < 4; ++t) {
                Frag B;
                B.q[0] = *(const u32x4*)(bbase[t] + jj);
                B.q[1] = *(const u32x4*)(bbase[t] + jj + 16);
                acc[t] = __builtin_amdgcn_wmma_f32_16x16x32_bf16(
                    false, P2.v, false, B.v, (short)0, acc[t], false, false);
            }
        }
        __syncthreads();                               // (3) pbuf consumed
    }

    // epilogue: /l, +bias, ELU, non-temporal store (write-once output)
    float linv[8];
#pragma unroll
    for (int r = 0; r < 8; ++r) {
        const float l = __shfl(l_run, hk * 8 + r, 32);
        linv[r] = 1.0f / l;
    }
#pragma unroll
    for (int t = 0; t < 4; ++t) {
        const int nn = f0 + t * 16 + lm;
        const float bv = bias[nn];
#pragma unroll
        for (int r = 0; r < 8; ++r) {
            float v = acc[t][r] * linv[r] + bv;
            v = v > 0.0f ? v : (__expf(v) - 1.0f);   // ELU
            __builtin_nontemporal_store(v, out + (size_t)(i0 + hk * 8 + r) * FOUT + nn);
        }
    }
}

// ---------------------------------------------------------------------------
// K4: RWR closed form. One block per row; count adj_ad in {2,3}.
// 256MB one-shot stream -> non-temporal loads.
// ---------------------------------------------------------------------------
__global__ void rwr_kernel(const int* __restrict__ adj_ad, float* __restrict__ ro) {
    const int i = blockIdx.x;
    const int* row = adj_ad + (size_t)i * NN;
    int cnt = 0;
    for (int j = threadIdx.x * 4; j < NN; j += blockDim.x * 4) {
        i32x4 v = __builtin_nontemporal_load((const i32x4*)(row + j));
#pragma unroll
        for (int p = 0; p < 4; ++p) cnt += (v[p] > 1 && v[p] < 4) ? 1 : 0;
    }
#pragma unroll
    for (int o = 16; o; o >>= 1) cnt += __shfl_down(cnt, o, 32);
    __shared__ int rc[8];
    if ((threadIdx.x & 31) == 0) rc[threadIdx.x >> 5] = cnt;
    __syncthreads();
    if (threadIdx.x == 0) {
        int tot = 0;
#pragma unroll
        for (int w = 0; w < 8; ++w) tot += rc[w];
        const float k = (float)tot;
        const float denom = 1.0f - 0.25f * k;      // 1 - c^2 k, c=0.5
        float ri0, rin;
        if (denom == 0.0f) { ri0 = 1.0f; rin = 0.5f; }
        else               { ri0 = fabsf(1.0f / denom); rin = fabsf(0.5f / denom); }
        __builtin_nontemporal_store(ri0, ro + i * 2);
        __builtin_nontemporal_store(rin, ro + i * 2 + 1);
    }
}

// ---------------------------------------------------------------------------
extern "C" void kernel_launch(void* const* d_in, const int* in_sizes, int n_in,
                              void* d_out, int out_size, void* d_ws, size_t ws_size,
                              hipStream_t stream) {
    (void)in_sizes; (void)n_in; (void)out_size; (void)ws_size;

    const float* x      = (const float*)d_in[0];
    const float* adj    = (const float*)d_in[1];
    const int*   adj_ad = (const int*)  d_in[2];
    const float* seq_W  = (const float*)d_in[3];
    const float* f1_w   = (const float*)d_in[4];
    const float* f1_b   = (const float*)d_in[5];
    const float* f2_w   = (const float*)d_in[6];
    const float* f2_b   = (const float*)d_in[7];
    const float* bias   = (const float*)d_in[8];
    float* out = (float*)d_out;

    char* ws = (char*)d_ws;
    const size_t XBF  = (size_t)NN * FIN * 2;     // 16 MB
    const size_t WBF  = (size_t)FOUT * FIN * 2;   //  1 MB
    const size_t SF32 = (size_t)NN * FOUT * 4;    // 16 MB
    const size_t SBFT = (size_t)FOUT * NN * 2;    //  8 MB
    unsigned short* xbf  = (unsigned short*)(ws);
    unsigned short* wbf  = (unsigned short*)(ws + XBF);
    float*          sf   = (float*)         (ws + XBF + WBF);
    unsigned short* sbfT = (unsigned short*)(ws + XBF + WBF + SF32);
    float*          f1   = (float*)         (ws + XBF + WBF + SF32 + SBFT);
    float*          f2   = f1 + NN;

    // K0: conversions
    cvt_f32_bf16<<<(NN * FIN) / 1024, 256, 0, stream>>>(x, xbf, NN * FIN);
    cvt_f32_bf16<<<(FOUT * FIN) / 1024, 256, 0, stream>>>(seq_W, wbf, FOUT * FIN);

    // K1: seq_fts GEMM (512 blocks x 8 waves, 16x64 tiles)
    gemm_seqfts<<<512, 256, 0, stream>>>(xbf, wbf, sf, sbfT);

    // K2: f1 / f2 row reductions
    f12_kernel<<<NN, 256, 0, stream>>>(sf, f1_w, f1_b, f2_w, f2_b, f1, f2);

    // K3: fused softmax-attention GEMM -> h_prime (ELU) into d_out[0 .. NN*FOUT)
    flash_attn<<<NN / 16, 256, 0, stream>>>(adj, f1, f2, sbfT, bias, out);

    // K4: RWR into d_out[NN*FOUT ..)
    rwr_kernel<<<NN, 256, 0, stream>>>(adj_ad, out + (size_t)NN * FOUT);
}
